// WindowAttention_7103875907881
// MI455X (gfx1250) — matrix operand order repaired and checked by reading, not compiled, not run on hardware
//
#include <hip/hip_runtime.h>

typedef __attribute__((ext_vector_type(16))) _Float16 v16h;
typedef __attribute__((ext_vector_type(8)))  _Float16 v8h;
typedef __attribute__((ext_vector_type(4)))  _Float16 v4h;
typedef __attribute__((ext_vector_type(8)))  float    v8f;
typedef __attribute__((ext_vector_type(4)))  float    v4f;

#define DIM     384
#define NHEADS  12
#define HD      32
#define WIN     64
#define QKSCALE 0.17677669529663687f   // (384/12)^-0.5

#define VT_STRIDE 72    // VhT row stride in halves (64 + 8 pad)
#define X_STRIDE  392   // Xbuf row stride in halves (384 + 8 pad)
#define P_STRIDE  72    // Pbuf row stride in halves

static __device__ inline v16h cat8(v8h a, v8h b) {
  return __builtin_shufflevector(a, b, 0,1,2,3,4,5,6,7,8,9,10,11,12,13,14,15);
}

static __device__ inline v8h cvt8(v4f a, v4f b, float s) {
  v8h r;
  r[0]=(_Float16)(a.x*s); r[1]=(_Float16)(a.y*s); r[2]=(_Float16)(a.z*s); r[3]=(_Float16)(a.w*s);
  r[4]=(_Float16)(b.x*s); r[5]=(_Float16)(b.y*s); r[6]=(_Float16)(b.z*s); r[7]=(_Float16)(b.w*s);
  return r;
}

static __device__ inline v8f wmma_f16(v16h a, v16h b, v8f c) {
  return __builtin_amdgcn_wmma_f32_16x16x32_f16(false, a, false, b, (short)0, c, false, false);
}

// One-shot f32 -> f16 conversion of proj_w into workspace (reused by all blocks via L2)
__global__ __launch_bounds__(256)
void cvt_w_kernel(const float* __restrict__ w, _Float16* __restrict__ wh) {
  int i = blockIdx.x * blockDim.x + threadIdx.x;   // over float4s: 384*384/4 = 36864
  v4f v = *(const v4f*)(w + (size_t)i * 4);
  v4h h; h[0]=(_Float16)v.x; h[1]=(_Float16)v.y; h[2]=(_Float16)v.z; h[3]=(_Float16)v.w;
  *(v4h*)(wh + (size_t)i * 4) = h;
}

__global__ __launch_bounds__(256)
void win_attn_kernel(const float* __restrict__ qkv,
                     const float* __restrict__ mask,
                     const _Float16* __restrict__ Wh,
                     const float* __restrict__ proj_b,
                     float* __restrict__ out) {
  // LDS: 55296 + 50176 + 16384 + 18432 = 140288 bytes
  __shared__ __align__(16) _Float16 VhT[DIM][VT_STRIDE];   // V transposed: [channel][token]
  __shared__ __align__(16) _Float16 Xbuf[WIN][X_STRIDE];   // attention output, A-operand of proj
  __shared__ __align__(16) float    maskS[WIN][WIN];
  __shared__ __align__(16) _Float16 Pbuf[8][16][P_STRIDE]; // per-wave softmax P bounce buffer

  const int win  = blockIdx.x;
  const int tid  = threadIdx.x;
  const int wave = tid >> 5;
  const int lane = tid & 31;
  const int l16  = lane & 15;
  const int hi   = lane >> 4;

  const float* qkv_w = qkv + (size_t)win * (WIN * 3 * DIM);

  // ---------------- Phase 0: stage V (transposed) and mask ----------------
  #pragma unroll
  for (int i = 0; i < 24; ++i) {
    int q   = tid + 256 * i;          // 0..6143 float4s of the V slice
    int tok = q / 96;
    int c   = (q % 96) * 4;
    v4f v = *(const v4f*)(qkv_w + (size_t)tok * (3 * DIM) + 2 * DIM + c);
    VhT[c + 0][tok] = (_Float16)v.x;
    VhT[c + 1][tok] = (_Float16)v.y;
    VhT[c + 2][tok] = (_Float16)v.z;
    VhT[c + 3][tok] = (_Float16)v.w;
  }
  const float* mask_w = mask + (size_t)(win & 63) * (WIN * WIN);
  #pragma unroll
  for (int i = 0; i < 4; ++i) {
    int q = tid + 256 * i;            // 0..1023 float4s
    ((v4f*)maskS)[q] = *(const v4f*)(mask_w + (size_t)q * 4);
  }
  __syncthreads();

  // ---------------- Phase 1: attention (48 head x M-tile tasks over 8 waves) ----------------
  for (int task = wave; task < 48; task += 8) {
    const int h  = task >> 2;
    const int mt = task & 3;
    const int M0 = mt * 16;
    const int hb = h * HD;

    // Q A-fragment straight from global (f32 -> f16, pre-scaled)
    const float* qrow = qkv_w + (size_t)(M0 + l16) * (3 * DIM) + hb;
    v4f q0 = *(const v4f*)(qrow + 16 * hi);
    v4f q1 = *(const v4f*)(qrow + 16 * hi + 4);
    v4f q2 = *(const v4f*)(qrow + 16 * hi + 8);
    v4f q3 = *(const v4f*)(qrow + 16 * hi + 12);
    v16h aq = cat8(cvt8(q0, q1, QKSCALE), cvt8(q2, q3, QKSCALE));

    // S = Q @ K^T : 4 N-tiles, K-dim = 32 channels (contiguous in qkv rows)
    v8f s[4];
    #pragma unroll
    for (int nt = 0; nt < 4; ++nt) {
      const float* krow = qkv_w + (size_t)(nt * 16 + l16) * (3 * DIM) + DIM + hb + 16 * hi;
      v4f k0 = *(const v4f*)(krow + 0);
      v4f k1 = *(const v4f*)(krow + 4);
      v4f k2 = *(const v4f*)(krow + 8);
      v4f k3 = *(const v4f*)(krow + 12);
      v16h bk = cat8(cvt8(k0, k1, 1.0f), cvt8(k2, k3, 1.0f));
      v8f z = {0.f,0.f,0.f,0.f,0.f,0.f,0.f,0.f};
      s[nt] = wmma_f16(aq, bk, z);
    }

    // + mask, then row softmax in registers (rows live across 16-lane halves)
    #pragma unroll
    for (int nt = 0; nt < 4; ++nt)
      #pragma unroll
      for (int r = 0; r < 8; ++r)
        s[nt][r] += maskS[M0 + r + 8 * hi][nt * 16 + l16];

    #pragma unroll
    for (int r = 0; r < 8; ++r) {
      float m = fmaxf(fmaxf(s[0][r], s[1][r]), fmaxf(s[2][r], s[3][r]));
      m = fmaxf(m, __shfl_xor(m, 1, 32));
      m = fmaxf(m, __shfl_xor(m, 2, 32));
      m = fmaxf(m, __shfl_xor(m, 4, 32));
      m = fmaxf(m, __shfl_xor(m, 8, 32));
      float sum = 0.f;
      #pragma unroll
      for (int nt = 0; nt < 4; ++nt) {
        float e = __expf(s[nt][r] - m);
        s[nt][r] = e;
        sum += e;
      }
      sum += __shfl_xor(sum, 1, 32);
      sum += __shfl_xor(sum, 2, 32);
      sum += __shfl_xor(sum, 4, 32);
      sum += __shfl_xor(sum, 8, 32);
      float inv = 1.0f / sum;
      #pragma unroll
      for (int nt = 0; nt < 4; ++nt)
        Pbuf[wave][r + 8 * hi][nt * 16 + l16] = (_Float16)(s[nt][r] * inv);
    }

    // O = P @ V : K-dim = 64 tokens (2 chunks), 2 N-tiles of 16 channels
    v8f o0 = {0.f,0.f,0.f,0.f,0.f,0.f,0.f,0.f};
    v8f o1 = {0.f,0.f,0.f,0.f,0.f,0.f,0.f,0.f};
    #pragma unroll
    for (int kc = 0; kc < 2; ++kc) {
      v8h plo = *(const v8h*)&Pbuf[wave][l16][kc * 32 + 16 * hi];
      v8h phi = *(const v8h*)&Pbuf[wave][l16][kc * 32 + 16 * hi + 8];
      v16h ap = cat8(plo, phi);
      v8h va0 = *(const v8h*)&VhT[hb + l16][kc * 32 + 16 * hi];
      v8h va1 = *(const v8h*)&VhT[hb + l16][kc * 32 + 16 * hi + 8];
      v8h vb0 = *(const v8h*)&VhT[hb + 16 + l16][kc * 32 + 16 * hi];
      v8h vb1 = *(const v8h*)&VhT[hb + 16 + l16][kc * 32 + 16 * hi + 8];
      o0 = wmma_f16(ap, cat8(va0, va1), o0);
      o1 = wmma_f16(ap, cat8(vb0, vb1), o1);
    }
    #pragma unroll
    for (int r = 0; r < 8; ++r) {
      Xbuf[M0 + r + 8 * hi][hb + l16]      = (_Float16)o0[r];
      Xbuf[M0 + r + 8 * hi][hb + 16 + l16] = (_Float16)o1[r];
    }
  }
  __syncthreads();

  // ---------------- Phase 2: projection Y = X @ W^T + b ----------------
  for (int nt = wave; nt < 24; nt += 8) {
    const int N0 = nt * 16;
    const float bias = proj_b[N0 + l16];
    v8f acc[4];
    #pragma unroll
    for (int mt = 0; mt < 4; ++mt)
      acc[mt] = (v8f){0.f,0.f,0.f,0.f,0.f,0.f,0.f,0.f};

    for (int kc = 0; kc < 12; ++kc) {
      const _Float16* wrow = Wh + (size_t)(N0 + l16) * DIM + kc * 32 + 16 * hi;
      v8h w0 = *(const v8h*)(wrow);
      v8h w1 = *(const v8h*)(wrow + 8);
      v16h bw = cat8(w0, w1);
      #pragma unroll
      for (int mt = 0; mt < 4; ++mt) {
        v8h x0 = *(const v8h*)&Xbuf[mt * 16 + l16][kc * 32 + 16 * hi];
        v8h x1 = *(const v8h*)&Xbuf[mt * 16 + l16][kc * 32 + 16 * hi + 8];
        acc[mt] = wmma_f16(cat8(x0, x1), bw, acc[mt]);
      }
    }
    #pragma unroll
    for (int mt = 0; mt < 4; ++mt)
      #pragma unroll
      for (int r = 0; r < 8; ++r) {
        int row = mt * 16 + r + 8 * hi;
        out[((size_t)win * 64 + row) * DIM + N0 + l16] = acc[mt][r] + bias;
      }
  }
}

extern "C" void kernel_launch(void* const* d_in, const int* in_sizes, int n_in,
                              void* d_out, int out_size, void* d_ws, size_t ws_size,
                              hipStream_t stream) {
  (void)n_in; (void)out_size; (void)ws_size;
  const float* qkv    = (const float*)d_in[0];
  const float* mask   = (const float*)d_in[1];
  const float* proj_w = (const float*)d_in[2];
  const float* proj_b = (const float*)d_in[3];
  float* out = (float*)d_out;
  _Float16* Wh = (_Float16*)d_ws;           // 384*384 f16 = 294912 B

  int B_ = in_sizes[0] / (WIN * 3 * DIM);   // 2048

  cvt_w_kernel<<<(DIM * DIM / 4) / 256, 256, 0, stream>>>(proj_w, Wh);
  win_attn_kernel<<<B_, 256, 0, stream>>>(qkv, mask, Wh, proj_b, out);
}
